// VladPoolingLayer_77232101916695
// MI455X (gfx1250) — matrix operand order repaired and checked by reading, not compiled, not run on hardware
//
#include <hip/hip_runtime.h>
#include <math.h>

typedef __attribute__((ext_vector_type(16))) __bf16 v16bf;
typedef __attribute__((ext_vector_type(8)))  __bf16 v8bf;
typedef __attribute__((ext_vector_type(8)))  float  v8f;

// Primary path: true CDNA5 async global->LDS copy via inline asm.
#define USE_ASYNC_ASM 1

#if defined(__has_builtin)
#  if __has_builtin(__builtin_amdgcn_load_to_lds)
#    define HAVE_LOAD_TO_LDS 1
#  endif
#endif
#ifndef HAVE_LOAD_TO_LDS
#  define HAVE_LOAD_TO_LDS 0
#endif

namespace {

constexpr int D    = 512;
constexpr int KH   = 7;
constexpr int K    = D * KH;     // 3584 (im2col depth)
constexpr int B    = 32;
constexpr int W    = 250;
constexpr int M    = B * W;      // 8000 (output positions)
constexpr int NCL  = 8;
constexpr int KG   = 10;

constexpr int BM = 128;
constexpr int BN = 128;
constexpr int BK = 32;

union Frag { v16bf v; v8bf h[2]; };

__device__ __forceinline__ unsigned int pack_bf16(float lo, float hi) {
    unsigned int ulo = __float_as_uint(lo);
    ulo += 0x7FFFu + ((ulo >> 16) & 1u);          // RNE round to bf16
    unsigned int uhi = __float_as_uint(hi);
    uhi += 0x7FFFu + ((uhi >> 16) & 1u);
    return (ulo >> 16) | (uhi & 0xFFFF0000u);
}

__device__ __forceinline__ unsigned short f2bf(float f) {
    unsigned int u = __float_as_uint(f);
    u += 0x7FFFu + ((u >> 16) & 1u);
    return (unsigned short)(u >> 16);
}

// 16-byte global -> LDS copy.
// USE_ASYNC_ASM: GLOBAL_LOAD_ASYNC_TO_LDS_B128 (ASYNCcnt-tracked DMA into LDS).
//   VDST = per-lane LDS byte address (low 32 bits of the generic pointer, per
//   ISA 10.2: LDS aperture address is addr[31:0]); VADDR = 64-bit global addr.
__device__ __forceinline__ void g2l16(const void* g, void* l) {
#if USE_ASYNC_ASM
    const unsigned int loff = (unsigned int)(unsigned long long)l;
    asm volatile("global_load_async_to_lds_b128 %0, %1, off"
                 :: "v"(loff), "v"(g) : "memory");
#elif HAVE_LOAD_TO_LDS
    __builtin_amdgcn_load_to_lds(
        (void*)g,
        (__attribute__((address_space(3))) void*)(unsigned int)(unsigned long long)l,
        16, 0, 0);
#else
    *(uint4*)l = *(const uint4*)g;
#endif
}

__device__ __forceinline__ void wait_async_lds() {
#if USE_ASYNC_ASM
    asm volatile("s_wait_asynccnt 0x0" ::: "memory");
#endif
}

// ---------------------------------------------------------------------------
// Prepass 1: im2col pack.  x is [b][K][W] (since c*(KH*W)+h*W == kg*W), so this
// is a per-b [K][W] -> [W][K] transpose with f32->bf16:
//   xp[b*W + w][kg] = bf16(x[b][kg][w])
// LDS-tiled: coalesced reads over w, coalesced 64B writes over k.
// ---------------------------------------------------------------------------
__global__ __launch_bounds__(256) void im2col_pack_k(
    const float* __restrict__ x, unsigned short* __restrict__ xp)
{
    __shared__ __align__(16) unsigned short tile[64][136];  // 128 + 8 pad (keeps 16B align)

    const int bb = blockIdx.z;
    const int w0 = blockIdx.x * 64;
    const int k0 = blockIdx.y * 128;
    const int tid = threadIdx.x;

    // read phase: thread (tw, kq) covers w = w0+tw, k = k0 + kq*32 .. +31
    const int tw = tid & 63;
    const int kq = tid >> 6;                        // 0..3
    const int wr = w0 + tw < W ? w0 + tw : W - 1;   // clamp (dup reads harmless)
    const float* xb = x + (long)bb * (K * W) + (long)(k0 + kq * 32) * W + wr;
    #pragma unroll
    for (int j = 0; j < 32; ++j) {
        tile[tw][kq * 32 + j] = f2bf(xb[(long)j * W]);
    }
    __syncthreads();

    // write phase: thread covers row tid>>2, 32 contiguous k (64 bytes)
    const int rw = tid >> 2;                 // 0..63
    const int ck = (tid & 3) * 32;           // 0,32,64,96
    const int gw = w0 + rw;
    if (gw < W) {
        const long row = (long)bb * W + gw;
        uint4* dst = (uint4*)(xp + row * K + k0 + ck);
        const uint4* src = (const uint4*)&tile[rw][ck];
        dst[0] = src[0]; dst[1] = src[1]; dst[2] = src[2]; dst[3] = src[3];
    }
}

// ---------------------------------------------------------------------------
// Prepass 2: weight rows f32 -> bf16 (rows >= valid_rows are zero-filled)
// ---------------------------------------------------------------------------
__global__ __launch_bounds__(256) void pack_rows_k(
    const float* __restrict__ src, unsigned short* __restrict__ dst,
    int total_rows, int valid_rows)
{
    const long idx = ((long)blockIdx.x * 256 + threadIdx.x) * 8;
    const long total = (long)total_rows * K;
    if (idx >= total) return;
    const int row = (int)(idx / K);
    float v[8];
    if (row < valid_rows) {
        const float4* p = (const float4*)(src + idx);
        const float4 a = p[0], b = p[1];
        v[0]=a.x; v[1]=a.y; v[2]=a.z; v[3]=a.w;
        v[4]=b.x; v[5]=b.y; v[6]=b.z; v[7]=b.w;
    } else {
        #pragma unroll
        for (int i = 0; i < 8; ++i) v[i] = 0.f;
    }
    uint4 o;
    o.x = pack_bf16(v[0], v[1]); o.y = pack_bf16(v[2], v[3]);
    o.z = pack_bf16(v[4], v[5]); o.w = pack_bf16(v[6], v[7]);
    *(uint4*)(dst + idx) = o;
}

// ---------------------------------------------------------------------------
// Kernel 1: feat = relu(xp @ wp^T + feat_b)  [M=8000, N=512, K=3584], bf16 WMMA.
// Double-buffered LDS fed by async global->LDS; one barrier per K-step.
// ---------------------------------------------------------------------------
__global__ __launch_bounds__(256) void feat_gemm_k(
    const unsigned short* __restrict__ xp, const unsigned short* __restrict__ wp,
    const float* __restrict__ bias, float* __restrict__ feat)
{
    __shared__ __align__(16) unsigned short As[2][BM][BK];
    __shared__ __align__(16) unsigned short Bs[2][BN][BK];

    const int tid  = threadIdx.x;
    const int lane = tid & 31;
    const int wid  = tid >> 5;
    const int sel  = lane >> 4;
    const int rl   = lane & 15;
    const int wm   = (wid & 3) * 32;   // 4 waves in M
    const int wn   = (wid >> 2) * 64;  // 2 waves in N

    const int m0 = blockIdx.x * BM;
    const int n0 = blockIdx.y * BN;

    // A loader: row tid&127, 16 contiguous k at (tid>>7)*16. Tail rows clamp to
    // M-1 (their outputs are masked at the store, so duplicated data is fine).
    const int a_row = tid & 127;
    const int a_k0  = (tid >> 7) * 16;
    const int gmr   = m0 + a_row < M ? m0 + a_row : M - 1;
    const unsigned short* agp = xp + (long)gmr * K + a_k0;

    // B loader: row tid>>1, 16 contiguous k at (tid&1)*16
    const int b_row = tid >> 1;
    const int b_k0  = (tid & 1) * 16;
    const unsigned short* bgp = wp + (long)(n0 + b_row) * K + b_k0;

    v8f acc[2][4] = {};
    constexpr int NT = K / BK;   // 112

    auto stage = [&](int kt, int buf) {
        g2l16(agp + kt,     &As[buf][a_row][a_k0]);
        g2l16(agp + kt + 8, &As[buf][a_row][a_k0 + 8]);
        g2l16(bgp + kt,     &Bs[buf][b_row][b_k0]);
        g2l16(bgp + kt + 8, &Bs[buf][b_row][b_k0 + 8]);
    };

    stage(0, 0);
    for (int t = 0; t < NT; ++t) {
        const int buf = t & 1;
        wait_async_lds();                // own async copies landed in LDS
        __syncthreads();                 // buf ready; everyone done reading buf^1
        if (t + 1 < NT) stage((t + 1) * BK, buf ^ 1);

        Frag a[2], b[4];
        #pragma unroll
        for (int i = 0; i < 2; ++i) {
            const unsigned short* rowp = &As[buf][wm + i*16 + rl][0];
            a[i].h[0] = *(const v8bf*)(rowp + sel*8);        // K 0..7 / 8..15
            a[i].h[1] = *(const v8bf*)(rowp + 16 + sel*8);   // K 16..23 / 24..31
        }
        #pragma unroll
        for (int j = 0; j < 4; ++j) {
            const unsigned short* rowp = &Bs[buf][wn + j*16 + rl][0];
            b[j].h[0] = *(const v8bf*)(rowp + sel*8);        // K 0..7 / 8..15
            b[j].h[1] = *(const v8bf*)(rowp + 16 + sel*8);   // K 16..23 / 24..31
        }
        #pragma unroll
        for (int i = 0; i < 2; ++i)
            #pragma unroll
            for (int j = 0; j < 4; ++j)
                acc[i][j] = __builtin_amdgcn_wmma_f32_16x16x32_bf16(
                    false, a[i].v, false, b[j].v, (short)0, acc[i][j], false, false);
    }

    #pragma unroll
    for (int i = 0; i < 2; ++i) {
        #pragma unroll
        for (int j = 0; j < 4; ++j) {
            const int n = n0 + wn + j*16 + rl;
            const float bv = bias[n];
            #pragma unroll
            for (int v = 0; v < 8; ++v) {
                const int m = m0 + wm + i*16 + v + sel*8;
                if (m < M) feat[(long)m * D + n] = fmaxf(acc[i][j][v] + bv, 0.f);
            }
        }
    }
}

// ---------------------------------------------------------------------------
// Kernel 2: assign logits (N padded 10->16, awp prepacked/zero-padded) + softmax
// ---------------------------------------------------------------------------
__global__ __launch_bounds__(256) void assign_gemm_k(
    const unsigned short* __restrict__ xp, const unsigned short* __restrict__ awp,
    const float* __restrict__ abias, float* __restrict__ soft)
{
    __shared__ __align__(16) unsigned short As[2][BM][BK];
    __shared__ __align__(16) unsigned short Bs[2][16][BK];
    __shared__ float lg[BM][17];

    const int tid  = threadIdx.x;
    const int lane = tid & 31;
    const int wid  = tid >> 5;
    const int sel  = lane >> 4;
    const int rl   = lane & 15;
    const int wm   = wid * 16;           // 8 waves, one 16-row M-tile each

    const int m0 = blockIdx.x * BM;

    const int a_row = tid & 127;
    const int a_k0  = (tid >> 7) * 16;
    const int gmr   = m0 + a_row < M ? m0 + a_row : M - 1;
    const unsigned short* agp = xp + (long)gmr * K + a_k0;

    const int b_row = tid >> 1;          // only tid<32 participate
    const int b_k0  = (tid & 1) * 16;
    const unsigned short* bgp = awp + (long)b_row * K + b_k0;

    v8f acc = {};
    constexpr int NT = K / BK;

    auto stage = [&](int kt, int buf) {
        g2l16(agp + kt,     &As[buf][a_row][a_k0]);
        g2l16(agp + kt + 8, &As[buf][a_row][a_k0 + 8]);
        if (tid < 32) {
            g2l16(bgp + kt,     &Bs[buf][b_row][b_k0]);
            g2l16(bgp + kt + 8, &Bs[buf][b_row][b_k0 + 8]);
        }
    };

    stage(0, 0);
    for (int t = 0; t < NT; ++t) {
        const int buf = t & 1;
        wait_async_lds();
        __syncthreads();
        if (t + 1 < NT) stage((t + 1) * BK, buf ^ 1);

        Frag a, b;
        const unsigned short* ra = &As[buf][wm + rl][0];
        a.h[0] = *(const v8bf*)(ra + sel*8);
        a.h[1] = *(const v8bf*)(ra + 16 + sel*8);
        const unsigned short* rb = &Bs[buf][rl][0];
        b.h[0] = *(const v8bf*)(rb + sel*8);
        b.h[1] = *(const v8bf*)(rb + 16 + sel*8);

        acc = __builtin_amdgcn_wmma_f32_16x16x32_bf16(
            false, a.v, false, b.v, (short)0, acc, false, false);
    }

    #pragma unroll
    for (int v = 0; v < 8; ++v) lg[wm + v + sel*8][rl] = acc[v];
    __syncthreads();

    if (tid < BM) {
        const int m = m0 + tid;
        if (m < M) {
            float l[KG];
            float mx = -1e30f;
            #pragma unroll
            for (int k = 0; k < KG; ++k) { l[k] = lg[tid][k] + abias[k]; mx = fmaxf(mx, l[k]); }
            float s = 0.f;
            #pragma unroll
            for (int k = 0; k < KG; ++k) { l[k] = __expf(l[k] - mx); s += l[k]; }
            const float inv = 1.f / s;
            #pragma unroll
            for (int k = 0; k < KG; ++k) soft[(long)m * 16 + k] = l[k] * inv;
        }
    }
}

// ---------------------------------------------------------------------------
// Kernel 3: agg = soft^T @ feat per batch; subtract centroid*s_sum; L2 norm;
// transposed store. One block per batch element.
// ---------------------------------------------------------------------------
__global__ __launch_bounds__(256) void vlad_finalize_k(
    const float* __restrict__ feat, const float* __restrict__ soft,
    const float* __restrict__ cent, float* __restrict__ out)
{
    __shared__ float sw[W][KG];
    __shared__ float ssum[KG];
    __shared__ float nrm[NCL];

    const int bb  = blockIdx.x;
    const int tid = threadIdx.x;

    for (int idx = tid; idx < W * KG; idx += 256) {
        const int w = idx / KG;
        const int k = idx - w * KG;
        sw[w][k] = soft[(long)(bb * W + w) * 16 + k];
    }
    if (tid < NCL) nrm[tid] = 0.f;
    __syncthreads();
    if (tid < KG) {
        float s = 0.f;
        for (int w = 0; w < W; ++w) s += sw[w][tid];
        ssum[tid] = s;
    }
    __syncthreads();

    const int d0 = tid;
    const int d1 = tid + 256;
    float acc0[KG] = {};
    float acc1[KG] = {};
    const float* fb = feat + (long)bb * W * D;
    for (int w = 0; w < W; ++w) {
        const float f0 = fb[(long)w * D + d0];
        const float f1 = fb[(long)w * D + d1];
        #pragma unroll
        for (int k = 0; k < KG; ++k) {
            const float s = sw[w][k];
            acc0[k] = fmaf(s, f0, acc0[k]);
            acc1[k] = fmaf(s, f1, acc1[k]);
        }
    }

    float r0[NCL], r1[NCL];
    #pragma unroll
    for (int k = 0; k < NCL; ++k) {
        r0[k] = acc0[k] - cent[d0 * KG + k] * ssum[k];
        r1[k] = acc1[k] - cent[d1 * KG + k] * ssum[k];
        atomicAdd(&nrm[k], r0[k]*r0[k] + r1[k]*r1[k]);
    }
    __syncthreads();
    #pragma unroll
    for (int k = 0; k < NCL; ++k) {
        const float inv = 1.f / fmaxf(sqrtf(nrm[k]), 1e-12f);
        out[((long)bb * NCL + k) * D + d0] = r0[k] * inv;
        out[((long)bb * NCL + k) * D + d1] = r1[k] * inv;
    }
}

} // namespace

extern "C" void kernel_launch(void* const* d_in, const int* in_sizes, int n_in,
                              void* d_out, int out_size, void* d_ws, size_t ws_size,
                              hipStream_t stream) {
    const float* x        = (const float*)d_in[0];
    const float* feat_w   = (const float*)d_in[1];
    const float* feat_b   = (const float*)d_in[2];
    const float* assign_w = (const float*)d_in[3];
    const float* assign_b = (const float*)d_in[4];
    const float* cent     = (const float*)d_in[5];
    float* out = (float*)d_out;

    // workspace layout (all 16B aligned: K*2 = 7168 bytes per row)
    unsigned short* xp  = (unsigned short*)d_ws;          // M  x K bf16  (57.3 MB)
    unsigned short* wp  = xp  + (size_t)M * K;            // D  x K bf16  ( 3.7 MB)
    unsigned short* awp = wp  + (size_t)D * K;            // 16 x K bf16  ( 0.1 MB)
    float* feat_ws = (float*)(awp + (size_t)16 * K);      // M x D f32    (16.4 MB)
    float* soft_ws = feat_ws + (size_t)M * D;             // M x 16 f32   ( 0.5 MB)

    im2col_pack_k<<<dim3((W + 63) / 64, K / 128, B), 256, 0, stream>>>(x, xp);

    {
        const long total = (long)D * K;
        pack_rows_k<<<(int)((total / 8 + 255) / 256), 256, 0, stream>>>(feat_w, wp, D, D);
    }
    {
        const long total = 16L * K;
        pack_rows_k<<<(int)((total / 8 + 255) / 256), 256, 0, stream>>>(assign_w, awp, 16, KG);
    }

    feat_gemm_k<<<dim3((M + BM - 1) / BM, D / BN), 256, 0, stream>>>(xp, wp, feat_b, feat_ws);
    assign_gemm_k<<<(M + BM - 1) / BM, 256, 0, stream>>>(xp, awp, assign_b, soft_ws);
    vlad_finalize_k<<<dim3(B), 256, 0, stream>>>(feat_ws, soft_ws, cent, out);
}
